// Model_48928267436601
// MI455X (gfx1250) — matrix operand order repaired and checked by reading, not compiled, not run on hardware
//
#include <hip/hip_runtime.h>
#include <hip/hip_bf16.h>
#include <math.h>

// ---------------- problem constants ----------------
#define BATCH 8192
#define LLEN  512
#define PLEN  96
#define NFREQ 2048          // FFT/2
#define NEXP  8             // total experts
#define NER   6             // RLinear experts
#define NRCOL (NER * PLEN)  // 576
#define NFT   (NFREQ / 64)  // 32 freq tiles for the gate GEMM

typedef __bf16 bf16_t;
typedef __attribute__((ext_vector_type(8)))  bf16_t v8bf;
typedef __attribute__((ext_vector_type(16))) bf16_t v16bf;
typedef __attribute__((ext_vector_type(8)))  float  v8f;
typedef __attribute__((ext_vector_type(4)))  unsigned int v4u;
typedef __attribute__((ext_vector_type(8)))  int v8i;
typedef __attribute__((ext_vector_type(4)))  int v4i;

#define WMMA_BF16(A, B, C) \
  __builtin_amdgcn_wmma_f32_16x16x32_bf16(false, (A), false, (B), (short)0, (C), false, false)

__device__ __forceinline__ v16bf make_frag(v8bf lo, v8bf hi) {
  return __builtin_shufflevector(lo, hi, 0, 1, 2, 3, 4, 5, 6, 7, 8, 9, 10, 11, 12, 13, 14, 15);
}

// ---------------- TDM: 2D tile (tile_d0 x tile_d1, 2-byte elems) global -> LDS ----------------
// D# per CDNA5 ISA 8.3/8.4. LDS padding: +4 DWORDs after every 16 DWORDs (64B row -> 80B stride)
// This toolchain uses the 6-arg builtin: (g0 v4u, g1 v8i, g2 v4i, g3 v4i, v8i, cpol)
__device__ __forceinline__ void tdm_load_2d(unsigned lds_off, const void* gptr,
                                            int tile_d0, int tile_d1, unsigned stride_elems) {
  const unsigned long long ga = (unsigned long long)(uintptr_t)gptr;
  v4u g0;
  g0.x = 1u;                                              // count=1, user descriptor
  g0.y = lds_off;                                         // LDS byte address
  g0.z = (unsigned)(ga & 0xFFFFFFFFu);                    // global_addr[31:0]
  g0.w = (unsigned)((ga >> 32) & 0x01FFFFFFu) | (2u << 30); // global_addr[56:32] | type=2
  v8i g1;
  g1[0] = (int)((1u << 16) |        // data_size = 1 -> 2 bytes
                (1u << 20) |        // pad_enable
                (3u << 22) |        // pad_interval: 16 DWORDs (64B)
                (3u << 25));        // pad_amount: 4 DWORDs (16B)
  g1[1] = (int)0xFFFF0000u;                               // tensor_dim0[15:0]=0xFFFF (no OOB)
  g1[2] = (int)0xFFFF7FFFu;                               // tensor_dim0[31:16] | tensor_dim1[15:0]
  g1[3] = (int)(0x00007FFFu | ((unsigned)tile_d0 << 16)); // tensor_dim1[31:16] | tile_dim0
  g1[4] = (int)((unsigned)tile_d1 & 0xFFFFu);             // tile_dim1 (tile_dim2 = 0)
  g1[5] = (int)stride_elems;                              // tensor_dim0_stride[31:0]
  g1[6] = 0;                                              // stride[47:32] | dim1_stride[15:0]
  g1[7] = 0;
  const v4i z4 = {0, 0, 0, 0};
  const v8i z8 = {0, 0, 0, 0, 0, 0, 0, 0};
  __builtin_amdgcn_tensor_load_to_lds(g0, g1, z4, z4, z8, 0);
}

// ---------------- 1) row stats + centered bf16 copy ----------------
__global__ __launch_bounds__(256) void stats_kernel(const float* __restrict__ x,
                                                    float* __restrict__ mu,
                                                    float* __restrict__ sd,
                                                    float* __restrict__ last,
                                                    bf16_t* __restrict__ x0) {
  const int b = blockIdx.x;
  const int t = threadIdx.x;
  const int lane = t & 31, wv = t >> 5;
  const float* row = x + (size_t)b * LLEN;
  float v0 = row[t], v1 = row[t + 256];
  float s = v0 + v1, s2 = v0 * v0 + v1 * v1;
#pragma unroll
  for (int mask = 1; mask < 32; mask <<= 1) {
    s  += __shfl_xor(s, mask, 32);
    s2 += __shfl_xor(s2, mask, 32);
  }
  __shared__ float ls[8], ls2[8];
  __shared__ float smu;
  if (lane == 0) { ls[wv] = s; ls2[wv] = s2; }
  __syncthreads();
  if (t == 0) {
    float S = 0.f, S2 = 0.f;
#pragma unroll
    for (int w = 0; w < 8; ++w) { S += ls[w]; S2 += ls2[w]; }
    float mean = S * (1.0f / LLEN);
    float var  = S2 * (1.0f / LLEN) - mean * mean;
    smu = mean;
    mu[b] = mean;
    sd[b] = sqrtf(var + 1e-5f);
    last[b] = row[LLEN - 1];
  }
  __syncthreads();
  const float mean = smu;
  x0[(size_t)b * LLEN + t]       = (bf16_t)(v0 - mean);
  x0[(size_t)b * LLEN + t + 256] = (bf16_t)(v1 - mean);
}

// ---------------- 2a) DFT twiddles: rows 0..2047 = cos, 2048..4095 = -sin, /sqrt(4096) ----------------
__global__ __launch_bounds__(256) void twiddle_kernel(bf16_t* __restrict__ Tw) {
  const int idx = blockIdx.x * 256 + threadIdx.x;   // < 4096*512
  const int n = idx >> 9;
  const int l = idx & 511;
  const int f = n & (NFREQ - 1);
  const int r = (f * l) & 4095;                     // exact phase mod 2*pi
  const float theta = (float)r * 1.5339807878856412e-3f;  // 2*pi/4096
  float sn, cs;
  __sincosf(theta, &sn, &cs);
  const float val = ((n < NFREQ) ? cs : -sn) * (1.0f / 64.0f);  // 1/sqrt(4096)
  Tw[idx] = (bf16_t)val;
}

// ---------------- 2b) gating weight transpose: WgT[f][e] ----------------
__global__ __launch_bounds__(256) void wgt_kernel(const float* __restrict__ Wg,
                                                  float* __restrict__ WgT) {
  const int f = blockIdx.x * 256 + threadIdx.x;     // < 2048
#pragma unroll
  for (int e = 0; e < NEXP; ++e) WgT[f * NEXP + e] = Wg[e * NFREQ + f];
}

// ---------------- 2c) RLinear weights to bf16 ----------------
__global__ __launch_bounds__(256) void wrcast_kernel(const float* __restrict__ Wr,
                                                     bf16_t* __restrict__ Wrb) {
  const int i = blockIdx.x * 256 + threadIdx.x;     // < 576*512
  Wrb[i] = (bf16_t)Wr[i];
}

// ---------------- 3) DFT GEMM via TDM-staged LDS + fused periodogram/gating ----------------
// Block = 128 batch rows x 64 freqs. Waves split rows -> B tile shared 8x through LDS.
// Double-buffered tensor_load_to_lds pipeline, s_wait_tensorcnt + barriers.
#define LDSPITCH 40   // 32 elems + TDM pad (4 DW) -> 80B stride, conflict-free ds_load_b128
__global__ __launch_bounds__(256) void gate_gemm_tdm_kernel(const bf16_t* __restrict__ x0,
                                                            const bf16_t* __restrict__ Tw,
                                                            const float* __restrict__ WgT,
                                                            float* __restrict__ gate_part,
                                                            float* __restrict__ s_part) {
  const int ft = blockIdx.x & (NFT - 1);      // freq tile (64 freqs)
  const int bt = blockIdx.x / NFT;            // batch tile (128 rows)
  const int f0 = ft * 64;
  const int b0 = bt * 128;
  const int t = threadIdx.x;
  const int lane = t & 31, wv = t >> 5;
  const int m = lane & 15, hk = lane >> 4;

  __shared__ bf16_t ldsA[2][128][LDSPITCH];
  __shared__ bf16_t ldsC[2][64][LDSPITCH];
  __shared__ bf16_t ldsS[2][64][LDSPITCH];

  const bf16_t* gA = x0 + (size_t)b0 * LLEN;
  const bf16_t* gC = Tw + (size_t)f0 * LLEN;
  const bf16_t* gS = Tw + (size_t)(NFREQ + f0) * LLEN;

  v8f ccos[4] = {}, csin[4] = {};

  if (wv == 0) {  // only wave 0 drives the TDM
    tdm_load_2d((unsigned)(uintptr_t)&ldsA[0][0][0], gA, 32, 128, LLEN);
    tdm_load_2d((unsigned)(uintptr_t)&ldsC[0][0][0], gC, 32, 64, LLEN);
    tdm_load_2d((unsigned)(uintptr_t)&ldsS[0][0][0], gS, 32, 64, LLEN);
  }

  for (int kt = 0; kt < 16; ++kt) {
    const int buf = kt & 1;
    if (wv == 0) {
      if (kt + 1 < 16) {
        const int nb = buf ^ 1;
        const size_t koff = (size_t)(kt + 1) * 32;
        tdm_load_2d((unsigned)(uintptr_t)&ldsA[nb][0][0], gA + koff, 32, 128, LLEN);
        tdm_load_2d((unsigned)(uintptr_t)&ldsC[nb][0][0], gC + koff, 32, 64, LLEN);
        tdm_load_2d((unsigned)(uintptr_t)&ldsS[nb][0][0], gS + koff, 32, 64, LLEN);
        __builtin_amdgcn_s_wait_tensorcnt(3);   // current stage done, next 3 in flight
      } else {
        __builtin_amdgcn_s_wait_tensorcnt(0);
      }
    }
    __syncthreads();  // stage kt visible to all waves

    // Issue ALL LDS fragment loads first, then drain with back-to-back WMMAs:
    // avoids an s_wait_dscnt 0 stall in front of every v_wmma.
    const bf16_t* rA = &ldsA[buf][wv * 16 + m][0];
    const v16bf a = make_frag(*(const v8bf*)(rA + hk * 16),
                              *(const v8bf*)(rA + hk * 16 + 8));
    v16bf bc[4], bs[4];
#pragma unroll
    for (int j = 0; j < 4; ++j) {
      const bf16_t* rC = &ldsC[buf][j * 16 + m][hk * 16];
      bc[j] = make_frag(*(const v8bf*)(rC), *(const v8bf*)(rC + 8));
      const bf16_t* rS = &ldsS[buf][j * 16 + m][hk * 16];
      bs[j] = make_frag(*(const v8bf*)(rS), *(const v8bf*)(rS + 8));
    }
#pragma unroll
    for (int j = 0; j < 4; ++j) {
      ccos[j] = WMMA_BF16(a, bc[j], ccos[j]);
      csin[j] = WMMA_BF16(a, bs[j], csin[j]);
    }
    __syncthreads();  // all waves done with buf before TDM overwrites it
  }

  // I = Re^2 + Im^2, fused gating contraction over this block's 64 freqs
  float pg[8][NEXP];
  float ps[8];
#pragma unroll
  for (int r = 0; r < 8; ++r) {
    ps[r] = 0.f;
#pragma unroll
    for (int e = 0; e < NEXP; ++e) pg[r][e] = 0.f;
  }
#pragma unroll
  for (int j = 0; j < 4; ++j) {
    const int f = f0 + j * 16 + m;
    const v8f wg = *(const v8f*)(WgT + f * NEXP);
#pragma unroll
    for (int r = 0; r < 8; ++r) {
      const float I = ccos[j][r] * ccos[j][r] + csin[j][r] * csin[j][r];
      ps[r] += I;
#pragma unroll
      for (int e = 0; e < NEXP; ++e) pg[r][e] += I * wg[e];
    }
  }
  // reduce freq direction: sum the 16 lanes of each half-wave
#pragma unroll
  for (int mask = 1; mask <= 8; mask <<= 1) {
#pragma unroll
    for (int r = 0; r < 8; ++r) {
      ps[r] += __shfl_xor(ps[r], mask, 32);
#pragma unroll
      for (int e = 0; e < NEXP; ++e) pg[r][e] += __shfl_xor(pg[r][e], mask, 32);
    }
  }
  // each wave owns its 16 rows exclusively -> direct partial stores, no atomics
  if (lane == 0 || lane == 16) {
    const int rowbase = b0 + wv * 16 + hk * 8;
    float* gp = gate_part + ((size_t)ft * BATCH + rowbase) * NEXP;
    float* sp = s_part + (size_t)ft * BATCH + rowbase;
#pragma unroll
    for (int r = 0; r < 8; ++r) {
      sp[r] = ps[r];
#pragma unroll
      for (int e = 0; e < NEXP; ++e) gp[r * NEXP + e] = pg[r][e];
    }
  }
}

// ---------------- 3b) reduce the 32 freq-tile partials ----------------
__global__ __launch_bounds__(256) void gate_reduce_kernel(const float* __restrict__ gp,
                                                          const float* __restrict__ sp,
                                                          float* __restrict__ gate_acc,
                                                          float* __restrict__ s_acc) {
  const int idx = blockIdx.x * 256 + threadIdx.x;  // < BATCH*NEXP
  float g = 0.f;
#pragma unroll 4
  for (int ftile = 0; ftile < NFT; ++ftile) g += gp[(size_t)ftile * BATCH * NEXP + idx];
  gate_acc[idx] = g;
  if (idx < BATCH) {
    float s = 0.f;
#pragma unroll 4
    for (int ftile = 0; ftile < NFT; ++ftile) s += sp[(size_t)ftile * BATCH + idx];
    s_acc[idx] = s;
  }
}

// ---------------- 4) RLinear GEMM: R[b, e*96+p] = sum_l (x-mu) * Wr ----------------
__global__ __launch_bounds__(256) void rlinear_gemm_kernel(const bf16_t* __restrict__ x0,
                                                           const bf16_t* __restrict__ Wrb,
                                                           float* __restrict__ R) {
  const int t = threadIdx.x;
  const int lane = t & 31;
  const int wg = blockIdx.x * 8 + (t >> 5);          // 0 .. 4607
  const int btile = wg / 9, ntile = wg % 9;
  const int b0 = btile * 16, n0 = ntile * 64;
  const int m = lane & 15, hk = lane >> 4;

  const bf16_t* rowA = x0 + (size_t)(b0 + m) * LLEN;
  v8f acc[4] = {};
#pragma unroll 1
  for (int kt = 0; kt < 16; ++kt) {
    const int k0 = kt * 32;
    __builtin_prefetch(rowA + k0 + 32, 0, 3);
    const v16bf a = make_frag(*(const v8bf*)(rowA + k0 + hk * 16),
                              *(const v8bf*)(rowA + k0 + hk * 16 + 8));
#pragma unroll
    for (int j = 0; j < 4; ++j) {
      const int ncol = n0 + j * 16 + m;
      const v16bf b = *(const v16bf*)(Wrb + (size_t)ncol * LLEN + k0 + hk * 16);
      acc[j] = WMMA_BF16(a, b, acc[j]);
    }
  }
  const int mbase = b0 + hk * 8;
#pragma unroll
  for (int j = 0; j < 4; ++j)
#pragma unroll
    for (int r = 0; r < 8; ++r)
      R[(size_t)(mbase + r) * NRCOL + n0 + j * 16 + m] = acc[j][r];
}

// ---------------- 5) gate normalize + top-2 softmax + expert combine ----------------
__global__ __launch_bounds__(128) void combine_kernel(const float* __restrict__ gate_acc,
                                                      const float* __restrict__ s_acc,
                                                      const float* __restrict__ bg,
                                                      const float* __restrict__ mu,
                                                      const float* __restrict__ sd,
                                                      const float* __restrict__ last,
                                                      const float* __restrict__ R,
                                                      const float* __restrict__ br,
                                                      float* __restrict__ out) {
  const int b = blockIdx.x;
  const int t = threadIdx.x;
  __shared__ float g[NEXP];
  __shared__ int i0s, i1s;
  __shared__ float w0s, w1s;
  if (t < NEXP) {
    float s = s_acc[b];
    s = (s == 0.0f) ? 1.0f : s;
    g[t] = gate_acc[(size_t)b * NEXP + t] / s + bg[t];
  }
  __syncthreads();
  if (t == 0) {
    int i0 = 0; float v0 = g[0];
#pragma unroll
    for (int e = 1; e < NEXP; ++e) if (g[e] > v0) { v0 = g[e]; i0 = e; }
    int i1 = (i0 == 0) ? 1 : 0; float v1 = g[i1];
#pragma unroll
    for (int e = 0; e < NEXP; ++e) if (e != i0 && g[e] > v1) { v1 = g[e]; i1 = e; }
    const float mx = fmaxf(v0, v1);
    const float e0 = __expf(v0 - mx), e1 = __expf(v1 - mx);
    const float inv = 1.0f / (e0 + e1);
    i0s = i0; i1s = i1; w0s = e0 * inv; w1s = e1 * inv;
  }
  __syncthreads();
  if (t < PLEN) {
    const float muv = mu[b], sdv = sd[b], lst = last[b];
    const float* Rb = R + (size_t)b * NRCOL;
    auto expert = [&](int e) -> float {
      if (e == 0) return muv;        // mean expert
      if (e == 1) return lst;        // naive expert
      const int n = (e - 2) * PLEN + t;
      return Rb[n] + br[n] * sdv + muv;  // RLinear with folded RevIN denorm
    };
    out[(size_t)b * PLEN + t] = w0s * expert(i0s) + w1s * expert(i1s);
  }
}

// ---------------- launcher ----------------
extern "C" void kernel_launch(void* const* d_in, const int* in_sizes, int n_in,
                              void* d_out, int out_size, void* d_ws, size_t ws_size,
                              hipStream_t stream) {
  const float* x  = (const float*)d_in[0];   // [B, L]
  const float* Wg = (const float*)d_in[1];   // [E, NFREQ]
  const float* bg = (const float*)d_in[2];   // [E]
  const float* Wr = (const float*)d_in[3];   // [ER, P, L]
  const float* br = (const float*)d_in[4];   // [ER, P]
  float* out = (float*)d_out;                // [B, P]

  char* ws = (char*)d_ws;
  size_t off = 0;
  auto wsalloc = [&](size_t bytes) -> void* {
    void* p = (void*)(ws + off);
    off += (bytes + 255) & ~(size_t)255;
    return p;
  };
  float*  mu        = (float*)wsalloc((size_t)BATCH * 4);
  float*  sd        = (float*)wsalloc((size_t)BATCH * 4);
  float*  last      = (float*)wsalloc((size_t)BATCH * 4);
  float*  s_acc     = (float*)wsalloc((size_t)BATCH * 4);
  float*  gate_acc  = (float*)wsalloc((size_t)BATCH * NEXP * 4);
  float*  WgT       = (float*)wsalloc((size_t)NFREQ * NEXP * 4);
  bf16_t* x0        = (bf16_t*)wsalloc((size_t)BATCH * LLEN * 2);
  bf16_t* Tw        = (bf16_t*)wsalloc((size_t)2 * NFREQ * LLEN * 2);
  bf16_t* Wrb       = (bf16_t*)wsalloc((size_t)NRCOL * LLEN * 2);
  float*  R         = (float*)wsalloc((size_t)BATCH * NRCOL * 4);
  float*  gate_part = (float*)wsalloc((size_t)NFT * BATCH * NEXP * 4);
  float*  s_part    = (float*)wsalloc((size_t)NFT * BATCH * 4);

  stats_kernel<<<BATCH, 256, 0, stream>>>(x, mu, sd, last, x0);
  twiddle_kernel<<<(2 * NFREQ * LLEN) / 256, 256, 0, stream>>>(Tw);
  wgt_kernel<<<NFREQ / 256, 256, 0, stream>>>(Wg, WgT);
  wrcast_kernel<<<(NRCOL * LLEN) / 256, 256, 0, stream>>>(Wr, Wrb);
  gate_gemm_tdm_kernel<<<NFT * (BATCH / 128), 256, 0, stream>>>(x0, Tw, WgT, gate_part, s_part);
  gate_reduce_kernel<<<(BATCH * NEXP) / 256, 256, 0, stream>>>(gate_part, s_part, gate_acc, s_acc);
  rlinear_gemm_kernel<<<(BATCH / 16) * 9 / 8, 256, 0, stream>>>(x0, Wrb, R);
  combine_kernel<<<BATCH, 128, 0, stream>>>(gate_acc, s_acc, bg, mu, sd, last, R, br, out);
}
